// RnnSenderReinforce_36799279792759
// MI455X (gfx1250) — matrix operand order repaired and checked by reading, not compiled, not run on hardware
//
#include <hip/hip_runtime.h>
#include <hip/hip_bf16.h>
#include <math.h>
#include <stdint.h>

// ---------------------------------------------------------------------------
// Problem constants (match the reference)
// ---------------------------------------------------------------------------
#define B_    512
#define IN_   256
#define E_    256
#define H_    512
#define V_    32000
#define T_    32
#define G3H_  1536          // 3*H
#define NT_   250           // V_ / 128 logits tiles along N

typedef __bf16 bf16_t;
typedef __attribute__((ext_vector_type(16))) __bf16 v16bf;
typedef __attribute__((ext_vector_type(8)))  __bf16 v8bf;
typedef __attribute__((ext_vector_type(8)))  float  v8f;

// ---------------------------------------------------------------------------
// WMMA fragment helpers (bf16 A/B, f32 accum), wave32, 16x16x32
//   C[m][n] = sum_k A[m][k] * B[n][k]   (i.e. A * B^T, both row-major)
// ---------------------------------------------------------------------------
__device__ __forceinline__ v16bf load_frag_a(const bf16_t* __restrict__ A,
                                             int lda, int mBase, int kBase) {
  const int lane = threadIdx.x & 31;
  const int row  = mBase + (lane & 15);
  const int k0   = kBase + ((lane >> 4) << 3);          // +0 or +8
  const bf16_t* p = A + (size_t)row * lda + k0;
  v8bf lo = *(const v8bf*)p;                            // K = k0 .. k0+7
  v8bf hi = *(const v8bf*)(p + 16);                     // K = k0+16 .. k0+23
  v16bf r;
#pragma unroll
  for (int i = 0; i < 8; ++i) { r[i] = lo[i]; r[i + 8] = hi[i]; }
  return r;
}

// A-fragment sourced from an LDS-staged chunk [64 rows][128 cols] bf16.
__device__ __forceinline__ v16bf load_frag_a_lds(const bf16_t* Ach,
                                                 int mLoc, int kLoc) {
  const int lane = threadIdx.x & 31;
  const int row  = mLoc + (lane & 15);
  const int k0   = kLoc + ((lane >> 4) << 3);           // +0 or +8
  const bf16_t* p = Ach + row * 128 + k0;
  v8bf lo = *(const v8bf*)p;                            // K = k0 .. k0+7
  v8bf hi = *(const v8bf*)(p + 16);                     // K = k0+16 .. k0+23
  v16bf r;
#pragma unroll
  for (int i = 0; i < 8; ++i) { r[i] = lo[i]; r[i + 8] = hi[i]; }
  return r;
}

__device__ __forceinline__ v16bf load_frag_b(const bf16_t* __restrict__ Bw,
                                             int ldb, int nBase, int kBase) {
  const int lane = threadIdx.x & 31;
  const int col  = nBase + (lane & 15);
  const int k0   = kBase + ((lane >> 4) << 3);          // +0 or +8
  const bf16_t* p = Bw + (size_t)col * ldb + k0;
  v8bf lo = *(const v8bf*)p;
  v8bf hi = *(const v8bf*)(p + 16);
  v16bf r;
#pragma unroll
  for (int i = 0; i < 8; ++i) { r[i] = lo[i]; r[i + 8] = hi[i]; }
  return r;
}

__device__ __forceinline__ v8f wmma_bf16(v16bf a, v16bf b, v8f c) {
  return __builtin_amdgcn_wmma_f32_16x16x32_bf16(false, a, false, b,
                                                 (short)0, c, false, false);
}

// ---------------------------------------------------------------------------
// Async global -> LDS copy (CDNA5, tracked on ASYNCcnt).
//   lds_off: 32-bit LDS byte address (low 32 bits of shared generic addr)
//   gaddr  : 64-bit global byte address
// ---------------------------------------------------------------------------
__device__ __forceinline__ void async_copy_b128(uint32_t lds_off, uint64_t gaddr) {
  asm volatile("global_load_async_to_lds_b128 %0, %1, off"
               :: "v"(lds_off), "v"(gaddr) : "memory");
}
__device__ __forceinline__ void wait_asynccnt0() {
  asm volatile("s_wait_asynccnt 0" ::: "memory");
}

// ---------------------------------------------------------------------------
// Counter-based RNG: splitmix64 -> uniform(0,1) -> gumbel
// ---------------------------------------------------------------------------
__device__ __forceinline__ float hash_u01(uint32_t step, uint32_t row, uint32_t col) {
  uint64_t z = ((uint64_t)step << 48) ^ ((uint64_t)row << 24) ^ (uint64_t)col;
  z += 0x9E3779B97F4A7C15ull;
  z = (z ^ (z >> 30)) * 0xBF58476D1CE4E5B9ull;
  z = (z ^ (z >> 27)) * 0x94D049BB133111EBull;
  z ^= z >> 31;
  return ((float)(uint32_t)(z >> 40) + 0.5f) * (1.0f / 16777216.0f);
}
__device__ __forceinline__ float gumbel_noise(uint32_t step, uint32_t row, uint32_t col) {
  float u = hash_u01(step, row, col);
  return -__logf(-__logf(u));
}

// ---------------------------------------------------------------------------
// fp32 -> bf16 conversion
// ---------------------------------------------------------------------------
__global__ __launch_bounds__(256) void cvt_f2bf(const float* __restrict__ src,
                                                bf16_t* __restrict__ dst, int n) {
  int i = blockIdx.x * 256 + threadIdx.x;
  if (i < n) dst[i] = (bf16_t)src[i];
}

__global__ __launch_bounds__(256) void init_inp(const float* __restrict__ sos,
                                                bf16_t* __restrict__ inpB) {
  int i = blockIdx.x * 256 + threadIdx.x;          // B_*E_ total
  inpB[i] = (bf16_t)sos[i & (E_ - 1)];
}

// ---------------------------------------------------------------------------
// Generic WMMA GEMM: C[M][N] = A[M][K] * B[N][K]^T + bias[N]
// Block: 256 threads = 8 waves in 2(M) x 4(N).  Block tile: 64 x 64.
// ---------------------------------------------------------------------------
__global__ __launch_bounds__(256) void wmma_gemm_bias(
    const bf16_t* __restrict__ A, const bf16_t* __restrict__ Bw,
    const float* __restrict__ bias, float* __restrict__ C,
    int M, int N, int K) {
  const int wave  = threadIdx.x >> 5;
  const int waveM = wave >> 2;                 // 0..1
  const int waveN = wave & 3;                  // 0..3
  const int mTile = blockIdx.y * 64 + waveM * 32;
  const int nTile = blockIdx.x * 64 + waveN * 16;

  v8f acc0 = {}; v8f acc1 = {};
  for (int k = 0; k < K; k += 32) {
    v16bf bf = load_frag_b(Bw, K, nTile, k);
    v16bf a0 = load_frag_a(A, K, mTile, k);
    v16bf a1 = load_frag_a(A, K, mTile + 16, k);
    acc0 = wmma_bf16(a0, bf, acc0);
    acc1 = wmma_bf16(a1, bf, acc1);
  }

  const int lane = threadIdx.x & 31;
  const int col  = nTile + (lane & 15);
  const int rowH = (lane >> 4) << 3;           // 0 or 8
  const float bv = bias ? bias[col] : 0.0f;
#pragma unroll
  for (int v = 0; v < 8; ++v) {
    C[(size_t)(mTile + rowH + v) * N + col]      = acc0[v] + bv;
    C[(size_t)(mTile + 16 + rowH + v) * N + col] = acc1[v] + bv;
  }
}

// ---------------------------------------------------------------------------
// GRU gates (elementwise): h' = (1-z)*n + z*h
// ---------------------------------------------------------------------------
__global__ __launch_bounds__(256) void gru_gates(
    const float* __restrict__ gi, const float* __restrict__ gh,
    float* __restrict__ h, bf16_t* __restrict__ hB) {
  int i = blockIdx.x * 256 + threadIdx.x;      // B_*H_
  int b = i >> 9, j = i & (H_ - 1);
  const float* Gi = gi + (size_t)b * G3H_;
  const float* Gh = gh + (size_t)b * G3H_;
  float r = 1.0f / (1.0f + __expf(-(Gi[j]        + Gh[j])));
  float z = 1.0f / (1.0f + __expf(-(Gi[H_ + j]   + Gh[H_ + j])));
  float n = tanhf(Gi[2 * H_ + j] + r * Gh[2 * H_ + j]);
  float hp = h[i];
  float hn = (1.0f - z) * n + z * hp;
  h[i]  = hn;
  hB[i] = (bf16_t)hn;
}

// ---------------------------------------------------------------------------
// Fused logits tile + online softmax/gumbel partials.
// Block tile: 64(M) x 128(N). 8 waves in 2x4; each wave 32x32 = 2x2 frags.
// A (h) tile is staged via async global->LDS copies in K-chunks of 128;
// WMMA A-fragments are then fed from LDS, B (W_out) streams global->VGPR.
// Partial record per (row, tileN): [m, s, t, gbest, glogit, gidx]
// ---------------------------------------------------------------------------
__global__ __launch_bounds__(256) void logits_partial(
    const bf16_t* __restrict__ hB, const bf16_t* __restrict__ Wb,
    const float* __restrict__ b_out, float* __restrict__ part, int step) {
  __shared__ __align__(16) bf16_t Ach[64 * 128];   // 16 KB A chunk
  __shared__ float tile[64][132];                  // 33.8 KB (+pad)
  __shared__ float red[6][64][4];                  // 6 KB

  const int wave  = threadIdx.x >> 5;
  const int lane  = threadIdx.x & 31;
  const int waveM = wave >> 2;                 // 0..1
  const int waveN = wave & 3;                  // 0..3
  const int mBlk  = blockIdx.y * 64;
  const int nBlk  = blockIdx.x * 128;
  const int n0    = nBlk + waveN * 32;
  const int mLoc  = waveM * 32;

  v8f acc00 = {}, acc01 = {}, acc10 = {}, acc11 = {};

  for (int kc = 0; kc < H_; kc += 128) {
    // --- async-stage A chunk: 64 rows x 128 bf16 = 1024 x 16B, 4 per thread
#pragma unroll
    for (int it = 0; it < 4; ++it) {
      int g  = threadIdx.x + it * 256;         // 0..1023
      int ar = g >> 4;                         // row 0..63
      int ac = g & 15;                         // 16B chunk 0..15
      uint32_t loff = (uint32_t)(uintptr_t)(Ach + ar * 128 + ac * 8);
      uint64_t gsrc = (uint64_t)(uintptr_t)(hB + (size_t)(mBlk + ar) * H_ +
                                            kc + ac * 8);
      async_copy_b128(loff, gsrc);
    }
    wait_asynccnt0();
    __syncthreads();

    // --- 4 WMMA K-steps over the staged chunk
#pragma unroll
    for (int kk = 0; kk < 128; kk += 32) {
      const int k = kc + kk;
      // hint the next L2-resident weight chunk
      if (k + 64 < H_)
        __builtin_prefetch(Wb + (size_t)(n0 + (lane & 15)) * H_ + k + 64, 0, 1);
      v16bf a0 = load_frag_a_lds(Ach, mLoc, kk);
      v16bf a1 = load_frag_a_lds(Ach, mLoc + 16, kk);
      v16bf b0 = load_frag_b(Wb, H_, n0, k);
      v16bf b1 = load_frag_b(Wb, H_, n0 + 16, k);
      acc00 = wmma_bf16(a0, b0, acc00);
      acc01 = wmma_bf16(a0, b1, acc01);
      acc10 = wmma_bf16(a1, b0, acc10);
      acc11 = wmma_bf16(a1, b1, acc11);
    }
    __syncthreads();   // all waves done reading before next chunk overwrites
  }

  // Spill the 32x32 wave tile (as 2x2 frags) to LDS with bias added.
  const int rowH = (lane >> 4) << 3;           // 0 or 8
  const int colL = lane & 15;
  {
    const int wr = waveM * 32, wc = waveN * 32;
#pragma unroll
    for (int v = 0; v < 8; ++v) {
      int r0 = wr + rowH + v, r1 = wr + 16 + rowH + v;
      tile[r0][wc + colL]      = acc00[v] + b_out[nBlk + wc + colL];
      tile[r0][wc + 16 + colL] = acc01[v] + b_out[nBlk + wc + 16 + colL];
      tile[r1][wc + colL]      = acc10[v] + b_out[nBlk + wc + colL];
      tile[r1][wc + 16 + colL] = acc11[v] + b_out[nBlk + wc + 16 + colL];
    }
  }
  __syncthreads();

  // Row-wise online stats: 4 threads per row, 32 cols each.
  const int r   = threadIdx.x >> 2;
  const int seg = threadIdx.x & 3;
  float m = -1e30f, s = 0.0f, t = 0.0f;
  float gb = -1e30f, gl = 0.0f; int gidx = 0;
#pragma unroll 4
  for (int c = seg * 32; c < seg * 32 + 32; ++c) {
    float l  = tile[r][c];
    float nm = fmaxf(m, l);
    float sc = __expf(m - nm);
    float e  = __expf(l - nm);
    s = s * sc + e;
    t = t * sc + l * e;
    m = nm;
    float g = l + gumbel_noise((uint32_t)step, (uint32_t)(mBlk + r),
                               (uint32_t)(nBlk + c));
    if (g > gb) { gb = g; gl = l; gidx = nBlk + c; }
  }
  red[0][r][seg] = m;  red[1][r][seg] = s;  red[2][r][seg] = t;
  red[3][r][seg] = gb; red[4][r][seg] = gl; red[5][r][seg] = __int_as_float(gidx);
  __syncthreads();

  if (seg == 0) {
    float M0 = red[0][r][0], S0 = red[1][r][0], Tt = red[2][r][0];
    float G0 = red[3][r][0], L0 = red[4][r][0]; int I0 = __float_as_int(red[5][r][0]);
#pragma unroll
    for (int q = 1; q < 4; ++q) {
      float m2 = red[0][r][q], s2 = red[1][r][q], t2 = red[2][r][q];
      float nm = fmaxf(M0, m2);
      float f1 = __expf(M0 - nm), f2 = __expf(m2 - nm);
      S0 = S0 * f1 + s2 * f2;
      Tt = Tt * f1 + t2 * f2;
      M0 = nm;
      if (red[3][r][q] > G0) {
        G0 = red[3][r][q]; L0 = red[4][r][q]; I0 = __float_as_int(red[5][r][q]);
      }
    }
    float* rec = part + ((size_t)(mBlk + r) * NT_ + blockIdx.x) * 6;
    rec[0] = M0; rec[1] = S0; rec[2] = Tt;
    rec[3] = G0; rec[4] = L0; rec[5] = __int_as_float(I0);
  }
}

// ---------------------------------------------------------------------------
// Per-row final reduction + stop head + sampling + emb gather.
// ---------------------------------------------------------------------------
__global__ __launch_bounds__(256) void reduce_sample(
    const float* __restrict__ part, const float* __restrict__ h,
    const float* __restrict__ W_stop, const float* __restrict__ b_stop,
    const float* __restrict__ emb, bf16_t* __restrict__ inpB,
    float* __restrict__ out, int step) {
  const int row = blockIdx.x;
  const int tid = threadIdx.x;

  __shared__ float sh[256];
  __shared__ float Rm[256], Rs[256], Rt[256], Rg[256], Rl[256];
  __shared__ int   Ri[256];
  __shared__ float sStop;
  __shared__ int   sSymb;

  // --- stop logit: dot(h[row], W_stop) over H_=512 ---
  {
    const float* hr = h + (size_t)row * H_;
    sh[tid] = hr[tid] * W_stop[tid] + hr[tid + 256] * W_stop[tid + 256];
  }
  __syncthreads();
  for (int o = 128; o > 0; o >>= 1) {
    if (tid < o) sh[tid] += sh[tid + o];
    __syncthreads();
  }
  if (tid == 0) sStop = sh[0] + b_stop[0];

  // --- merge 250 tile partials ---
  float m = -1e30f, s = 0.0f, t = 0.0f;
  float gb = -1e30f, gl = 0.0f; int gidx = 0;
  for (int tn = tid; tn < NT_; tn += 256) {
    const float* rec = part + ((size_t)row * NT_ + tn) * 6;
    float m2 = rec[0], s2 = rec[1], t2 = rec[2];
    float nm = fmaxf(m, m2);
    float f1 = __expf(m - nm), f2 = __expf(m2 - nm);
    s = s * f1 + s2 * f2;
    t = t * f1 + t2 * f2;
    m = nm;
    if (rec[3] > gb) { gb = rec[3]; gl = rec[4]; gidx = __float_as_int(rec[5]); }
  }
  Rm[tid] = m; Rs[tid] = s; Rt[tid] = t; Rg[tid] = gb; Rl[tid] = gl; Ri[tid] = gidx;
  __syncthreads();
  for (int o = 128; o > 0; o >>= 1) {
    if (tid < o) {
      float m2 = Rm[tid + o], s2 = Rs[tid + o], t2 = Rt[tid + o];
      float nm = fmaxf(Rm[tid], m2);
      float f1 = __expf(Rm[tid] - nm), f2 = __expf(m2 - nm);
      Rs[tid] = Rs[tid] * f1 + s2 * f2;
      Rt[tid] = Rt[tid] * f1 + t2 * f2;
      Rm[tid] = nm;
      if (Rg[tid + o] > Rg[tid]) {
        Rg[tid] = Rg[tid + o]; Rl[tid] = Rl[tid + o]; Ri[tid] = Ri[tid + o];
      }
    }
    __syncthreads();
  }

  if (tid == 0) {
    float logZ = Rm[0] + __logf(Rs[0]);
    float ent  = logZ - Rt[0] / Rs[0];
    int   symb = Ri[0];
    float symb_lp = Rl[0] - logZ;

    float sl = sStop;
    float p  = 1.0f / (1.0f + __expf(-sl));
    float u  = hash_u01((uint32_t)step, (uint32_t)row, 0x7FFFFFu);
    float stop = (u < p) ? 1.0f : 0.0f;
    float sp = (sl > 0.0f) ? (sl + log1pf(__expf(-sl))) : log1pf(__expf(sl));
    float stop_lp  = stop * sl - sp;
    float stop_ent = sp - sl * p;

    const int idx = row * T_ + step;
    ((int*)out)[0 * (B_ * T_) + idx] = symb;          // symb_seq (int)
    ((int*)out)[1 * (B_ * T_) + idx] = (int)stop;     // stop_seq (int32)
    out[2 * (B_ * T_) + idx] = symb_lp;
    out[3 * (B_ * T_) + idx] = stop_lp;
    out[4 * (B_ * T_) + idx] = ent;
    out[5 * (B_ * T_) + idx] = stop_ent;
    sSymb = symb;
  }
  __syncthreads();

  // --- next input: emb[symb] -> bf16 (E_ == 256 == blockDim) ---
  inpB[(size_t)row * E_ + tid] = (bf16_t)emb[(size_t)sSymb * E_ + tid];
}

// ---------------------------------------------------------------------------
// Launcher
// ---------------------------------------------------------------------------
extern "C" void kernel_launch(void* const* d_in, const int* in_sizes, int n_in,
                              void* d_out, int out_size, void* d_ws, size_t ws_size,
                              hipStream_t stream) {
  const float* x       = (const float*)d_in[0];
  const float* agent_w = (const float*)d_in[1];
  const float* agent_b = (const float*)d_in[2];
  const float* sos     = (const float*)d_in[3];
  const float* emb     = (const float*)d_in[4];
  const float* W_ih    = (const float*)d_in[5];
  const float* W_hh    = (const float*)d_in[6];
  const float* b_ih    = (const float*)d_in[7];
  const float* b_hh    = (const float*)d_in[8];
  const float* W_out   = (const float*)d_in[9];
  const float* b_out   = (const float*)d_in[10];
  const float* W_stop  = (const float*)d_in[11];
  const float* b_stop  = (const float*)d_in[12];

  char* ws = (char*)d_ws;
  size_t off = 0;
  auto alloc = [&](size_t bytes) -> char* {
    char* p = ws + off;
    off = (off + bytes + 255) & ~(size_t)255;
    return p;
  };
  bf16_t* WoutB = (bf16_t*)alloc((size_t)V_ * H_ * 2);
  bf16_t* WihB  = (bf16_t*)alloc((size_t)G3H_ * E_ * 2);
  bf16_t* WhhB  = (bf16_t*)alloc((size_t)G3H_ * H_ * 2);
  bf16_t* AwB   = (bf16_t*)alloc((size_t)H_ * IN_ * 2);
  bf16_t* xB    = (bf16_t*)alloc((size_t)B_ * IN_ * 2);
  float*  hBuf  = (float*)alloc((size_t)B_ * H_ * 4);
  bf16_t* hB    = (bf16_t*)alloc((size_t)B_ * H_ * 2);
  bf16_t* inpB  = (bf16_t*)alloc((size_t)B_ * E_ * 2);
  float*  gi    = (float*)alloc((size_t)B_ * G3H_ * 4);
  float*  gh    = (float*)alloc((size_t)B_ * G3H_ * 4);
  float*  part  = (float*)alloc((size_t)B_ * NT_ * 6 * 4);
  (void)ws_size; (void)in_sizes; (void)n_in; (void)out_size;

  auto cvt = [&](const float* src, bf16_t* dst, int n) {
    cvt_f2bf<<<(n + 255) / 256, 256, 0, stream>>>(src, dst, n);
  };
  cvt(W_out,   WoutB, V_ * H_);
  cvt(W_ih,    WihB,  G3H_ * E_);
  cvt(W_hh,    WhhB,  G3H_ * H_);
  cvt(agent_w, AwB,   H_ * IN_);
  cvt(x,       xB,    B_ * IN_);

  // h0 = x @ agent_w.T + agent_b
  wmma_gemm_bias<<<dim3(H_ / 64, B_ / 64), 256, 0, stream>>>(
      xB, AwB, agent_b, hBuf, B_, H_, IN_);
  cvt(hBuf, hB, B_ * H_);
  init_inp<<<(B_ * E_) / 256, 256, 0, stream>>>(sos, inpB);

  float* out = (float*)d_out;
  for (int t = 0; t < T_; ++t) {
    // gi = inp @ W_ih.T + b_ih   (K = E_)
    wmma_gemm_bias<<<dim3(G3H_ / 64, B_ / 64), 256, 0, stream>>>(
        inpB, WihB, b_ih, gi, B_, G3H_, E_);
    // gh = h @ W_hh.T + b_hh     (K = H_)
    wmma_gemm_bias<<<dim3(G3H_ / 64, B_ / 64), 256, 0, stream>>>(
        hB, WhhB, b_hh, gh, B_, G3H_, H_);
    // gate fusion -> new h (fp32 + bf16)
    gru_gates<<<(B_ * H_) / 256, 256, 0, stream>>>(gi, gh, hBuf, hB);
    // logits tiles + softmax/gumbel partials (async LDS staging + WMMA)
    logits_partial<<<dim3(NT_, B_ / 64), 256, 0, stream>>>(
        hB, WoutB, b_out, part, t);
    // final per-row reduce, sampling, stop head, emb gather
    reduce_sample<<<B_, 256, 0, stream>>>(
        part, hBuf, W_stop, b_stop, emb, inpB, out, t);
  }
}
